// Forwarder_79817672228879
// MI455X (gfx1250) — matrix operand order repaired and checked by reading, not compiled, not run on hardware
//
#include <hip/hip_runtime.h>

typedef __attribute__((ext_vector_type(16))) __bf16 v16bf;
typedef __attribute__((ext_vector_type(8)))  float  v8f;

#define VCB 128
#define BS 4096
#define LDST 130            // padded LDS row stride (ushorts) to break bank conflicts
#define BSVCB ((size_t)BS * VCB)

__device__ __forceinline__ unsigned short f32_to_bf16(float f) {
    unsigned u = __builtin_bit_cast(unsigned, f);
    unsigned r = u + 0x7FFFu + ((u >> 16) & 1u);   // round-to-nearest-even
    return (unsigned short)(r >> 16);
}
__device__ __forceinline__ float bf16_to_f32(unsigned short h) {
    unsigned u = ((unsigned)h) << 16;
    return __builtin_bit_cast(float, u);
}

// block-wide reductions over 128 values (all 256 threads participate in barriers)
__device__ __forceinline__ float blk_max(float v, float* red, int tid) {
    if (tid < VCB) red[tid] = v;
    __syncthreads();
    for (int s = 64; s > 0; s >>= 1) {
        if (tid < s) red[tid] = fmaxf(red[tid], red[tid + s]);
        __syncthreads();
    }
    float r = red[0];
    __syncthreads();
    return r;
}
__device__ __forceinline__ float blk_sum(float v, float* red, int tid) {
    if (tid < VCB) red[tid] = v;
    __syncthreads();
    for (int s = 64; s > 0; s >>= 1) {
        if (tid < s) red[tid] = red[tid] + red[tid + s];
        __syncthreads();
    }
    float r = red[0];
    __syncthreads();
    return r;
}

__global__ __launch_bounds__(256) void fwd_kernel(
    const int* __restrict__ x, const int* __restrict__ t,
    const float* __restrict__ lpr, const float* __restrict__ gumbel,
    const float* __restrict__ trans_logits, const float* __restrict__ prior_logits,
    float* __restrict__ out)
{
    extern __shared__ char smem[];
    unsigned short* MT = (unsigned short*)smem;          // [VCB][LDST] bf16, M transposed (col-major)
    unsigned short* Pm = MT + VCB * LDST;                // [VCB][LDST] bf16, P row-major
    float* sample  = (float*)(Pm + VCB * LDST);
    float* mprior  = sample  + VCB;
    float* pprobs  = mprior  + VCB;
    float* pmprobs = pprobs  + VCB;
    float* red     = pmprobs + VCB;                      // [128] reduction scratch

    const int b    = blockIdx.x;
    const int tid  = threadIdx.x;
    const int lane = tid & 31;
    const int wave = tid >> 5;
    const int half = lane >> 4;
    const int lm   = lane & 15;

    const int tb = t[b];
    const int xb = x[b];

    if (tid < VCB) { pprobs[tid] = 0.0f; pmprobs[tid] = 0.0f; }

    // ---- stage M (128x128 f32) -> LDS as bf16, transposed ----
    // trans_logits reads stay RT: the unique (t,x) working set (<=128 MB) fits the 192 MB L2.
    const float* Mg = trans_logits + ((size_t)(tb * 8 + xb)) * (VCB * VCB);
    #pragma unroll
    for (int it = 0; it < 16; ++it) {
        int idx4 = tid + it * 256;                       // float4 index, coalesced
        const float4 v = ((const float4*)Mg)[idx4];
        int e = idx4 * 4;
        int r = e >> 7;
        int c = e & 127;
        MT[(c + 0) * LDST + r] = f32_to_bf16(v.x);
        MT[(c + 1) * LDST + r] = f32_to_bf16(v.y);
        MT[(c + 2) * LDST + r] = f32_to_bf16(v.z);
        MT[(c + 3) * LDST + r] = f32_to_bf16(v.w);
    }

    // ---- vector softmaxes: prior/sample from lpr+gumbel, mprior from prior_logits[t] ----
    float lpv = -1e30f, gv = 0.0f, lmv = -1e30f;
    if (tid < VCB) {
        lpv = lpr[(size_t)b * VCB + tid];
        gv  = gumbel[(size_t)b * VCB + tid];
        lmv = prior_logits[(size_t)tb * VCB + tid];
        __builtin_nontemporal_store(lpv, &out[3 * BSVCB + (size_t)b * VCB + tid]); // lpr passthrough
        __builtin_nontemporal_store(lmv, &out[2 * BSVCB + (size_t)b * VCB + tid]); // lmpr
    }
    float mx = blk_max(lpv, red, tid);
    float e0 = (tid < VCB) ? __expf(lpv - mx) : 0.0f;
    float sm = blk_sum(e0, red, tid);
    float prior = e0 / sm;
    float sl = (tid < VCB) ? (__logf(prior + 1e-20f) + gv) * 0.5f : -1e30f;
    mx = blk_max(sl, red, tid);
    float e1 = (tid < VCB) ? __expf(sl - mx) : 0.0f;
    sm = blk_sum(e1, red, tid);
    if (tid < VCB) sample[tid] = e1 / sm;
    mx = blk_max(lmv, red, tid);
    float e2 = (tid < VCB) ? __expf(lmv - mx) : 0.0f;
    sm = blk_sum(e2, red, tid);
    if (tid < VCB) mprior[tid] = e2 / sm;

    // ---- P = softmax(M / 0.2) row-wise -> bf16 in LDS (thread = row) ----
    if (tid < VCB) {
        const int r = tid;
        float rmx = -1e30f;
        for (int c = 0; c < VCB; ++c)
            rmx = fmaxf(rmx, bf16_to_f32(MT[c * LDST + r]));
        float rs = 0.0f;
        for (int c = 0; c < VCB; ++c) {
            float e = __expf((bf16_to_f32(MT[c * LDST + r]) - rmx) * 5.0f);
            rs += e;
            Pm[r * LDST + c] = f32_to_bf16(e);
        }
        float inv = 1.0f / rs;
        for (int c = 0; c < VCB; ++c)
            Pm[r * LDST + c] = f32_to_bf16(bf16_to_f32(Pm[r * LDST + c]) * inv);
    }
    __syncthreads();

    // ---- trans_matrix = P @ M via v_wmma_f32_16x16x32_bf16 ----
    // wave w owns rows [16w,16w+16); 8 N-tiles x 4 K-steps = 32 WMMAs
    v8f acc[8] = {};
    const int arow = 16 * wave + lm;                      // A: M = lane&15 (both halves)
    for (int kc = 0; kc < 4; ++kc) {
        const int kb = kc * 32;
        union { v16bf v; unsigned short s[16]; } A;
        #pragma unroll
        for (int j = 0; j < 8; ++j) {                     // ISA 16-bit A 16x32 layout
            A.s[j]     = Pm[arow * LDST + kb + half * 8 + j];
            A.s[j + 8] = Pm[arow * LDST + kb + 16 + half * 8 + j];
        }
        #pragma unroll
        for (int n = 0; n < 8; ++n) {
            union { v16bf v; unsigned short s[16]; } B;
            const int bcol = 16 * n + lm;                 // ISA 16-bit B 32x16 layout
            #pragma unroll
            for (int j = 0; j < 16; ++j)
                B.s[j] = MT[bcol * LDST + kb + (j >> 3) * 16 + half * 8 + (j & 7)];
            acc[n] = __builtin_amdgcn_wmma_f32_16x16x32_bf16(
                false, A.v, false, B.v, (short)0, acc[n], false, false);
        }
    }

    // ---- store trans_matrix (f32) with NT hint: 256 MB streaming write, never re-read;
    //      keep L2 free for the trans_logits working set ----
    float* out_tm = out + 4 * BSVCB + (size_t)b * VCB * VCB;
    #pragma unroll
    for (int n = 0; n < 8; ++n) {
        #pragma unroll
        for (int r = 0; r < 8; ++r) {
            int row = 16 * wave + r + half * 8;
            int col = 16 * n + lm;
            __builtin_nontemporal_store(acc[n][r], &out_tm[row * VCB + col]);
        }
    }

    // ---- T = softmax(rows) in registers; partial sample@T, mprior@T ----
    float pp[8] = {}, pm[8] = {};
    #pragma unroll
    for (int r = 0; r < 8; ++r) {
        float rmx = -1e30f;
        #pragma unroll
        for (int n = 0; n < 8; ++n) rmx = fmaxf(rmx, acc[n][r]);
        for (int msk = 1; msk <= 8; msk <<= 1)            // reduce across 16-lane half
            rmx = fmaxf(rmx, __shfl_xor(rmx, msk, 32));
        float te[8];
        float rs = 0.0f;
        #pragma unroll
        for (int n = 0; n < 8; ++n) { te[n] = __expf(acc[n][r] - rmx); rs += te[n]; }
        for (int msk = 1; msk <= 8; msk <<= 1)
            rs += __shfl_xor(rs, msk, 32);
        float inv = 1.0f / rs;
        int row = 16 * wave + r + half * 8;
        float sv = sample[row];
        float mv = mprior[row];
        #pragma unroll
        for (int n = 0; n < 8; ++n) {
            float Tv = te[n] * inv;
            pp[n] += sv * Tv;
            pm[n] += mv * Tv;
        }
    }
    #pragma unroll
    for (int n = 0; n < 8; ++n) {
        float a = pp[n] + __shfl_xor(pp[n], 16, 32);      // merge row halves (same column)
        float c = pm[n] + __shfl_xor(pm[n], 16, 32);
        if (half == 0) {
            atomicAdd(&pprobs[16 * n + lm], a);           // ds_add_f32 across 8 waves
            atomicAdd(&pmprobs[16 * n + lm], c);
        }
    }
    __syncthreads();

    if (tid < VCB) {
        __builtin_nontemporal_store(pmprobs[tid], &out[0 * BSVCB + (size_t)b * VCB + tid]); // mprobs
        __builtin_nontemporal_store(pprobs[tid],  &out[1 * BSVCB + (size_t)b * VCB + tid]); // probs
    }
}

extern "C" void kernel_launch(void* const* d_in, const int* in_sizes, int n_in,
                              void* d_out, int out_size, void* d_ws, size_t ws_size,
                              hipStream_t stream) {
    const int*   x            = (const int*)d_in[0];
    const int*   t            = (const int*)d_in[1];
    const float* lpr          = (const float*)d_in[2];
    const float* gumbel       = (const float*)d_in[3];
    const float* trans_logits = (const float*)d_in[4];
    const float* prior_logits = (const float*)d_in[5];
    float* out = (float*)d_out;

    const size_t shmem = (size_t)2 * VCB * LDST * sizeof(unsigned short)
                       + (size_t)5 * VCB * sizeof(float);   // ~69 KB, fits 320 KB WGP LDS
    fwd_kernel<<<BS, 256, shmem, stream>>>(x, t, lpr, gumbel, trans_logits, prior_logits, out);
}
